// BinaryConv2d_75007308857785
// MI455X (gfx1250) — matrix operand order repaired and checked by reading, not compiled, not run on hardware
//
#include <hip/hip_runtime.h>

typedef __attribute__((ext_vector_type(16))) __bf16 v16bf;
typedef __attribute__((ext_vector_type(8)))  float  v8f;

#define NCH   16
#define IMH   256
#define IMW   256
#define NIMG  64
#define PLANE (IMH*IMW)                 // 65536
#define CNT_PER_CH (NIMG*PLANE)         // 4194304

// workspace layout (float offsets)
#define WS_ALPHA 0     // [16] per-out-channel alpha
#define WS_SUM   16    // [16] sum(y)
#define WS_SQ    32    // [16] sum(y^2)
#define WS_P     48    // [16] folded scale
#define WS_Q     64    // [16] folded bias
#define WS_ASWZ  80    // 5*32*16 bf16 = 2560 bf16 = 1280 floats (swizzled A operand)

// ---------------------------------------------------------------------------
// k0: alpha, swizzled sign-weight A operands, zero the accumulators
// A-operand lane layout for v_wmma_*_16x16x32_bf16 (ISA 7.12.2):
//   lane L holds row M = L%16; elem e -> K-chunk offset:
//   lane<16 : {0..7,16..23}; lane>=16 : {8..15,24..31}
// We instead use contiguous k = 16*(L>=16) + e with K = tap*16 + in_channel, so within chunk k:
//   t = 2k + (L>=16), c = e
// ---------------------------------------------------------------------------
__global__ void k0_prep(const float* __restrict__ rw, float* __restrict__ ws)
{
    int tid = threadIdx.x;
    if (tid < 16) {
        float s = 0.f;
        for (int i = 0; i < 144; ++i) s += fabsf(rw[tid * 144 + i]);
        ws[WS_ALPHA + tid] = s * (1.0f / 144.0f);
        ws[WS_SUM + tid]   = 0.f;
        ws[WS_SQ  + tid]   = 0.f;
    }
    __bf16* aswz = (__bf16*)(ws + WS_ASWZ);
    for (int idx = tid; idx < 5 * 32 * 16; idx += blockDim.x) {
        int e = idx & 15;
        int L = (idx >> 4) & 31;
        int k = idx >> 9;
        int M = L & 15;
        int t = 2 * k + ((L >= 16) ? 1 : 0);
        int c = e;
        float v = 0.f;
        if (t < 9) {
            float wv = rw[M * 144 + c * 9 + t];          // OIHW, t = kh*3+kw
            v = (wv > 0.f) ? 1.f : ((wv < 0.f) ? -1.f : 0.f);
        }
        aswz[idx] = (__bf16)v;
    }
}

// ---------------------------------------------------------------------------
// One 16(out-ch) x 16(pixels along w) tile of the binary conv via WMMA.
// B operand (32x16 bf16): lane group (lane/16) selects tap 2k / 2k+1,
// the 16 elems are the 16 input channels; column N = lane%16 = pixel.
//
// Branch-free B build: always load from a CLAMPED (legal) address, then
// zero out-of-range / padded-tap contributions with a mask multiply.
// This keeps all 16 strided loads unconditional -> fully pipelined vmem,
// one wait per chunk, EXEC stays all-ones for WMMA.
// ---------------------------------------------------------------------------
__device__ __forceinline__ v8f conv_tile(const float* __restrict__ xn,
                                         int h, int w0, int lane,
                                         const v16bf* __restrict__ a)
{
    v8f acc = {};
    const int group = lane >> 4;
    const int col   = lane & 15;
#pragma unroll
    for (int k = 0; k < 5; ++k) {
        int t = 2 * k + group;              // tap for this half-wave
        bool tv = (t < 9);
        int tt = tv ? t : 0;
        int dh = tt / 3 - 1;
        int dw = tt - (tt / 3) * 3 - 1;
        int hh = h + dh;
        int ww = w0 + col + dw;
        float mask = (tv && hh >= 0 && hh < IMH && ww >= 0 && ww < IMW)
                         ? 1.0f : 0.0f;
        int hc = hh < 0 ? 0 : (hh > IMH - 1 ? IMH - 1 : hh);
        int wc = ww < 0 ? 0 : (ww > IMW - 1 ? IMW - 1 : ww);
        const float* p = xn + hc * IMW + wc;
        v16bf b;
#pragma unroll
        for (int c = 0; c < 16; ++c) {
            float v = p[c * PLANE] * mask;  // unconditional load, masked value
            b[c] = (__bf16)v;
        }
        acc = __builtin_amdgcn_wmma_f32_16x16x32_bf16(
                  false, a[k], false, b, (short)0, acc, false, false);
    }
    return acc;
}

// ---------------------------------------------------------------------------
// k1: conv + per-channel sum / sumsq accumulation
// D layout: VGPR j, lane L -> M(out-ch) = j + 8*(L/16), N(pixel) = L%16
// ---------------------------------------------------------------------------
__global__ __launch_bounds__(256) void k1_stats(const float* __restrict__ x,
                                                float* __restrict__ ws)
{
    __shared__ float ssum[16], ssq[16];
    int tid = threadIdx.x;
    if (tid < 16) { ssum[tid] = 0.f; ssq[tid] = 0.f; }
    __syncthreads();

    int lane  = tid & 31;
    int group = lane >> 4;
    int gw = (int)(blockIdx.x * blockDim.x + tid) >> 5;
    int nw = (int)(gridDim.x * blockDim.x) >> 5;

    v16bf a[5];
    const v16bf* aswz = (const v16bf*)(ws + WS_ASWZ);
#pragma unroll
    for (int k = 0; k < 5; ++k) a[k] = aswz[k * 32 + lane];

    float af[8];
#pragma unroll
    for (int j = 0; j < 8; ++j) af[j] = ws[WS_ALPHA + j + 8 * group];

    float s[8], q[8];
#pragma unroll
    for (int j = 0; j < 8; ++j) { s[j] = 0.f; q[j] = 0.f; }

    const int totalT = NIMG * IMH * (IMW / 16);   // 262144
    for (int T = gw; T < totalT; T += nw) {
        int w0 = (T & 15) << 4;
        int h  = (T >> 4) & 255;
        int n  = T >> 12;
        v8f c = conv_tile(x + n * (NCH * PLANE), h, w0, lane, a);
#pragma unroll
        for (int j = 0; j < 8; ++j) {
            float y = c[j] * af[j];
            s[j] += y;
            q[j] += y * y;
        }
    }
    // reduce over the 16 pixel-lanes of each half-wave
#pragma unroll
    for (int m = 1; m <= 8; m <<= 1) {
#pragma unroll
        for (int j = 0; j < 8; ++j) {
            s[j] += __shfl_xor(s[j], m, 32);
            q[j] += __shfl_xor(q[j], m, 32);
        }
    }
    if ((lane & 15) == 0) {
#pragma unroll
        for (int j = 0; j < 8; ++j) {
            atomicAdd(&ssum[j + 8 * group], s[j]);
            atomicAdd(&ssq [j + 8 * group], q[j]);
        }
    }
    __syncthreads();
    if (tid < 16) {
        atomicAdd(&ws[WS_SUM + tid], ssum[tid]);
        atomicAdd(&ws[WS_SQ  + tid], ssq[tid]);
    }
}

// ---------------------------------------------------------------------------
// k2: finalize BN stats, fold alpha/gamma/beta into per-channel affine
// ---------------------------------------------------------------------------
__global__ void k2_finalize(const float* __restrict__ gamma,
                            const float* __restrict__ beta,
                            float* __restrict__ ws)
{
    int tid = threadIdx.x;
    if (tid < 16) {
        const float inv_cnt = 1.0f / (float)CNT_PER_CH;
        float mean = ws[WS_SUM + tid] * inv_cnt;
        float var  = ws[WS_SQ + tid] * inv_cnt - mean * mean;
        float istd = rsqrtf(var + 1e-5f);
        float g    = gamma[tid] * istd;
        ws[WS_P + tid] = ws[WS_ALPHA + tid] * g;   // applied to raw accumulator
        ws[WS_Q + tid] = beta[tid] - mean * g;
    }
}

// ---------------------------------------------------------------------------
// k3: recompute conv, apply folded BN affine, sign, store
// ---------------------------------------------------------------------------
__global__ __launch_bounds__(256) void k3_out(const float* __restrict__ x,
                                              const float* __restrict__ ws,
                                              float* __restrict__ out)
{
    int tid   = threadIdx.x;
    int lane  = tid & 31;
    int group = lane >> 4;
    int col   = lane & 15;
    int gw = (int)(blockIdx.x * blockDim.x + tid) >> 5;
    int nw = (int)(gridDim.x * blockDim.x) >> 5;

    v16bf a[5];
    const v16bf* aswz = (const v16bf*)(ws + WS_ASWZ);
#pragma unroll
    for (int k = 0; k < 5; ++k) a[k] = aswz[k * 32 + lane];

    float pf[8], qf[8];
#pragma unroll
    for (int j = 0; j < 8; ++j) {
        pf[j] = ws[WS_P + j + 8 * group];
        qf[j] = ws[WS_Q + j + 8 * group];
    }

    const int totalT = NIMG * IMH * (IMW / 16);
    for (int T = gw; T < totalT; T += nw) {
        int w0 = (T & 15) << 4;
        int h  = (T >> 4) & 255;
        int n  = T >> 12;
        v8f c = conv_tile(x + n * (NCH * PLANE), h, w0, lane, a);
#pragma unroll
        for (int j = 0; j < 8; ++j) {
            float v = c[j] * pf[j] + qf[j];
            float sg = (v > 0.f) ? 1.f : ((v < 0.f) ? -1.f : 0.f);
            int M = j + 8 * group;
            out[((n * NCH + M) * IMH + h) * IMW + w0 + col] = sg;
        }
    }
}

// ---------------------------------------------------------------------------
extern "C" void kernel_launch(void* const* d_in, const int* in_sizes, int n_in,
                              void* d_out, int out_size, void* d_ws, size_t ws_size,
                              hipStream_t stream)
{
    const float* x     = (const float*)d_in[0];
    const float* rw    = (const float*)d_in[1];
    const float* gamma = (const float*)d_in[2];
    const float* beta  = (const float*)d_in[3];
    float* out = (float*)d_out;
    float* ws  = (float*)d_ws;

    k0_prep    <<<dim3(1),    dim3(256), 0, stream>>>(rw, ws);
    k1_stats   <<<dim3(2048), dim3(256), 0, stream>>>(x, ws);
    k2_finalize<<<dim3(1),    dim3(32),  0, stream>>>(gamma, beta, ws);
    k3_out     <<<dim3(2048), dim3(256), 0, stream>>>(x, ws, out);
}
